// CrossAttention_26302379721473
// MI455X (gfx1250) — compile-verified, matrix operand order checked by reading, not run on hardware
//
#include <hip/hip_runtime.h>
#include <hip/hip_bf16.h>
#include <math.h>

// ---------------------------------------------------------------------------
// CrossAttention (self-attention) fused bf16-WMMA implementation for gfx1250.
// b=4, n=2560, qd=320, heads=8, dim_head=64, inner=512.
// Pipeline: convert -> QKV gemm (WMMA) -> flash attention (WMMA) -> out proj.
// GEMM waves compute 32x64 tiles (2 A-frags x 4 B-frags -> 8 WMMA / K-step)
// with all fragment loads hoisted ahead of the WMMA burst.
// ---------------------------------------------------------------------------

#define BATCH 4
#define SEQ   2560
#define QD    320
#define HEADS 8
#define DHEAD 64
#define INNER 512          // HEADS * DHEAD
#define ATT_SCALE 0.125f   // 1/sqrt(64)

typedef __bf16 bf16_t;
typedef __attribute__((ext_vector_type(16))) __bf16    v16bf;
typedef __attribute__((ext_vector_type(8)))  float     v8f;
typedef __attribute__((ext_vector_type(8)))  unsigned  v8u;

__device__ __forceinline__ bf16_t f2bf(float f) {
  unsigned u = __builtin_bit_cast(unsigned, f);
  u = u + 0x7FFFu + ((u >> 16) & 1u);                  // round-to-nearest-even
  unsigned short s = (unsigned short)(u >> 16);
  return __builtin_bit_cast(bf16_t, s);
}

// A-style 16x32 bf16 fragment load from row-major memory (row stride in elems).
// Also used for B fragments by passing the *transposed* operand (column-major
// load of B == row-major load of B^T), per ISA 7.12.2 layout:
//   lane = {half, m}:  VGPR g holds K = (g>=4?16:0) + 8*half + (g%4)*2 , +1
__device__ __forceinline__ v16bf load_frag(const bf16_t* __restrict__ base, int stride) {
  const int lane = threadIdx.x & 31;
  const int half = lane >> 4;
  const int m    = lane & 15;
  const unsigned* rp = (const unsigned*)(base + (size_t)m * stride);
  v8u r;
#pragma unroll
  for (int g = 0; g < 8; ++g) {
    const int k = ((g & 4) << 2) + half * 8 + (g & 3) * 2;
    r[g] = rp[k >> 1];
  }
  return __builtin_bit_cast(v16bf, r);
}

__device__ __forceinline__ v8f wmma_bf16(v16bf a, v16bf b, v8f c) {
  return __builtin_amdgcn_wmma_f32_16x16x32_bf16(false, a, false, b, (short)0, c,
                                                 false, false);
}

__device__ __forceinline__ float hmax16(float v) {
#pragma unroll
  for (int m = 1; m <= 8; m <<= 1) v = fmaxf(v, __shfl_xor(v, m, 32));
  return v;
}
__device__ __forceinline__ float hsum16(float v) {
#pragma unroll
  for (int m = 1; m <= 8; m <<= 1) v += __shfl_xor(v, m, 32);
  return v;
}

// --------------------------- conversion kernels ----------------------------

__global__ void k_cvt(const float* __restrict__ in, bf16_t* __restrict__ out, int n) {
  int i = blockIdx.x * blockDim.x + threadIdx.x;
  if (i < n) out[i] = f2bf(in[i]);
}

// in: [K][N] f32 row-major  ->  out: [N][K] bf16 (transpose)
__global__ void k_cvt_tr(const float* __restrict__ in, bf16_t* __restrict__ out,
                         int K, int N) {
  int i = blockIdx.x * blockDim.x + threadIdx.x;
  if (i < K * N) {
    int kk = i / N, n = i - kk * N;
    out[(size_t)n * K + kk] = f2bf(in[i]);
  }
}

// ------------------------------ QKV GEMM -----------------------------------
// C[M=10240, N=512] = xbf[10240,320] @ W.  Block: 128 thr = 4 waves, block
// tile 128(M)x64(N); each wave computes 32x64 via 2 A-frags x 4 B-frags.
// sel 0: Q (scaled, row-major)  sel 1: K (row-major)  sel 2: V (transposed).

__global__ __launch_bounds__(128) void k_qkv(const bf16_t* __restrict__ xbf,
                                             const bf16_t* __restrict__ WqT,
                                             const bf16_t* __restrict__ WkT,
                                             const bf16_t* __restrict__ WvT,
                                             bf16_t* __restrict__ Qb,
                                             bf16_t* __restrict__ Kb,
                                             bf16_t* __restrict__ Vt) {
  const int wave = threadIdx.x >> 5;
  const int lane = threadIdx.x & 31;
  const int half = lane >> 4;
  const int c    = lane & 15;
  const int m0   = blockIdx.x * 128 + wave * 32;  // flat (b*seq) row base
  const int n0   = blockIdx.y * 64;               // inner channel base
  const int sel  = blockIdx.z;
  const bf16_t* WT = (sel == 0) ? WqT : ((sel == 1) ? WkT : WvT);

  v8f acc[2][4] = {};
  for (int ks = 0; ks < QD / 32; ++ks) {
    const int k0 = ks * 32;
    v16bf a[2], b[4];
#pragma unroll
    for (int mi = 0; mi < 2; ++mi)
      a[mi] = load_frag(xbf + (size_t)(m0 + mi * 16) * QD + k0, QD);
#pragma unroll
    for (int t = 0; t < 4; ++t)
      b[t] = load_frag(WT + (size_t)(n0 + t * 16) * QD + k0, QD);
#pragma unroll
    for (int mi = 0; mi < 2; ++mi)
#pragma unroll
      for (int t = 0; t < 4; ++t)
        acc[mi][t] = wmma_bf16(a[mi], b[t], acc[mi][t]);
  }

  // Epilogue: uniform dispatch on sel; no per-element division (a 128-row
  // M-tile never straddles a batch boundary since 2560 % 128 == 0).
  if (sel == 0) {
#pragma unroll
    for (int mi = 0; mi < 2; ++mi)
#pragma unroll
      for (int t = 0; t < 4; ++t)
#pragma unroll
        for (int r = 0; r < 8; ++r) {
          const int row = m0 + mi * 16 + r + 8 * half;
          const int col = n0 + t * 16 + c;
          Qb[(size_t)row * INNER + col] = f2bf(acc[mi][t][r] * ATT_SCALE);
        }
  } else if (sel == 1) {
#pragma unroll
    for (int mi = 0; mi < 2; ++mi)
#pragma unroll
      for (int t = 0; t < 4; ++t)
#pragma unroll
        for (int r = 0; r < 8; ++r) {
          const int row = m0 + mi * 16 + r + 8 * half;
          const int col = n0 + t * 16 + c;
          Kb[(size_t)row * INNER + col] = f2bf(acc[mi][t][r]);
        }
  } else {
    const int b2   = m0 / SEQ;          // wave-uniform batch index
    const int tok0 = m0 - b2 * SEQ;     // wave-uniform token base
    bf16_t* Vb = Vt + (size_t)b2 * HEADS * DHEAD * SEQ;
#pragma unroll
    for (int mi = 0; mi < 2; ++mi)
#pragma unroll
      for (int t = 0; t < 4; ++t)
#pragma unroll
        for (int r = 0; r < 8; ++r) {
          const int tok = tok0 + mi * 16 + r + 8 * half;
          const int col = n0 + t * 16 + c;        // = head*64 + d
          Vb[(size_t)col * SEQ + tok] = f2bf(acc[mi][t][r]);
        }
  }
}

// --------------------------- flash attention -------------------------------
// grid (SEQ/64, BATCH*HEADS), block 128 = 4 waves; wave owns 16 query rows.
// Per 32-key tile: S = Q.K^T (2 WMMA-tiles), online softmax, P staged via LDS
// (bf16), O += P.V (4 WMMAs, V pre-transposed [b][h][d][n]).

__global__ __launch_bounds__(128) void k_attn(const bf16_t* __restrict__ Qb,
                                              const bf16_t* __restrict__ Kb,
                                              const bf16_t* __restrict__ Vt,
                                              bf16_t* __restrict__ Ob) {
  __shared__ __align__(16) bf16_t Pl[4][16 * 32];

  const int wave = threadIdx.x >> 5;
  const int lane = threadIdx.x & 31;
  const int half = lane >> 4;
  const int c    = lane & 15;
  const int b    = blockIdx.y >> 3;
  const int head = blockIdx.y & 7;
  const int q0   = blockIdx.x * 64 + wave * 16;

  const bf16_t* Qp = Qb + ((size_t)(b * SEQ + q0)) * INNER + head * DHEAD;
  const bf16_t* Kp = Kb + ((size_t)(b * SEQ)) * INNER + head * DHEAD;
  const bf16_t* Vp = Vt + ((size_t)(b * HEADS + head)) * DHEAD * SEQ;

  v16bf qf[2];
  qf[0] = load_frag(Qp, INNER);
  qf[1] = load_frag(Qp + 32, INNER);

  v8f o[4] = {};
  float mrow[8], lrow[8];
#pragma unroll
  for (int r = 0; r < 8; ++r) { mrow[r] = -3.0e38f; lrow[r] = 0.0f; }

  for (int kt = 0; kt < SEQ / 32; ++kt) {
    const int key0 = kt * 32;
    // prefetch next key tile (speculative; dropped past the end)
    __builtin_prefetch(Kp + (size_t)(key0 + 32) * INNER, 0, 1);
    __builtin_prefetch(Vp + (size_t)(lane & 15) * SEQ + key0 + 32, 0, 1);

    // ---- S = Q.K^T : load all 4 K fragments, then 4 WMMAs back-to-back
    v16bf kf[4];
#pragma unroll
    for (int j = 0; j < 2; ++j) {
      kf[j]     = load_frag(Kp + (size_t)key0 * INNER + j * 32, INNER);
      kf[2 + j] = load_frag(Kp + (size_t)(key0 + 16) * INNER + j * 32, INNER);
    }
    v8f s0 = {}, s1 = {};
    s0 = wmma_bf16(qf[0], kf[0], s0);
    s1 = wmma_bf16(qf[0], kf[2], s1);
    s0 = wmma_bf16(qf[1], kf[1], s0);
    s1 = wmma_bf16(qf[1], kf[3], s1);

    // ---- online softmax; C-layout: VGPR r <-> row (r + 8*half), col = c
#pragma unroll
    for (int r = 0; r < 8; ++r) {
      float t = hmax16(fmaxf(s0[r], s1[r]));
      const float mnew = fmaxf(mrow[r], t);
      const float corr = __expf(mrow[r] - mnew);
      mrow[r] = mnew;
      const float p0 = __expf(s0[r] - mnew);
      const float p1 = __expf(s1[r] - mnew);
      s0[r] = p0; s1[r] = p1;
      lrow[r] = lrow[r] * corr + hsum16(p0 + p1);
#pragma unroll
      for (int t2 = 0; t2 < 4; ++t2) o[t2][r] *= corr;
    }

    // ---- stage P (16q x 32k) in LDS bf16: C-tile -> A-fragment re-layout
#pragma unroll
    for (int r = 0; r < 8; ++r) {
      Pl[wave][(r + 8 * half) * 32 + c]      = f2bf(s0[r]);
      Pl[wave][(r + 8 * half) * 32 + 16 + c] = f2bf(s1[r]);
    }
    __syncthreads();
    v8u praw;
    {
      const int m = lane & 15;
      const unsigned* pw = (const unsigned*)(&Pl[wave][0]);
#pragma unroll
      for (int g = 0; g < 8; ++g) {
        const int k = ((g & 4) << 2) + half * 8 + (g & 3) * 2;
        praw[g] = pw[(m * 32 + k) >> 1];
      }
    }
    const v16bf pf = __builtin_bit_cast(v16bf, praw);

    // ---- O += P.V : load all 4 V fragments, then 4 WMMAs
    v16bf vf[4];
#pragma unroll
    for (int t2 = 0; t2 < 4; ++t2)
      vf[t2] = load_frag(Vp + (size_t)(t2 * 16) * SEQ + key0, SEQ);
#pragma unroll
    for (int t2 = 0; t2 < 4; ++t2)
      o[t2] = wmma_bf16(pf, vf[t2], o[t2]);
    __syncthreads();
  }
#pragma unroll
  for (int r = 0; r < 8; ++r) lrow[r] = 1.0f / lrow[r];
#pragma unroll
  for (int t2 = 0; t2 < 4; ++t2)
#pragma unroll
    for (int r = 0; r < 8; ++r) {
      const int row = q0 + r + 8 * half;
      const int col = head * DHEAD + t2 * 16 + c;
      Ob[((size_t)(b * SEQ + row)) * INNER + col] = f2bf(o[t2][r] * lrow[r]);
    }
}

// --------------------------- output projection -----------------------------
// out[10240,320] f32 = Ob[10240,512] @ Wo[512,320] + bo
// Block tile 128(M)x64(N), wave tile 32x64 (2 A-frags x 4 B-frags).

__global__ __launch_bounds__(128) void k_oproj(const bf16_t* __restrict__ Ob,
                                               const bf16_t* __restrict__ WoT,
                                               const float* __restrict__ bo,
                                               float* __restrict__ out) {
  const int wave = threadIdx.x >> 5;
  const int lane = threadIdx.x & 31;
  const int half = lane >> 4;
  const int c    = lane & 15;
  const int m0   = blockIdx.x * 128 + wave * 32;
  const int n0   = blockIdx.y * 64;

  v8f acc[2][4] = {};
  for (int ks = 0; ks < INNER / 32; ++ks) {
    const int k0 = ks * 32;
    v16bf a[2], b[4];
#pragma unroll
    for (int mi = 0; mi < 2; ++mi)
      a[mi] = load_frag(Ob + (size_t)(m0 + mi * 16) * INNER + k0, INNER);
#pragma unroll
    for (int t = 0; t < 4; ++t)
      b[t] = load_frag(WoT + (size_t)(n0 + t * 16) * INNER + k0, INNER);
#pragma unroll
    for (int mi = 0; mi < 2; ++mi)
#pragma unroll
      for (int t = 0; t < 4; ++t)
        acc[mi][t] = wmma_bf16(a[mi], b[t], acc[mi][t]);
  }
#pragma unroll
  for (int mi = 0; mi < 2; ++mi)
#pragma unroll
    for (int t = 0; t < 4; ++t)
#pragma unroll
      for (int r = 0; r < 8; ++r) {
        const int row = m0 + mi * 16 + r + 8 * half;
        const int col = n0 + t * 16 + c;
        out[(size_t)row * QD + col] = acc[mi][t][r] + bo[col];
      }
}

// ------------------------------- launcher ----------------------------------

extern "C" void kernel_launch(void* const* d_in, const int* in_sizes, int n_in,
                              void* d_out, int out_size, void* d_ws, size_t ws_size,
                              hipStream_t stream) {
  const float* x  = (const float*)d_in[0];
  const float* Wq = (const float*)d_in[1];
  const float* Wk = (const float*)d_in[2];
  const float* Wv = (const float*)d_in[3];
  const float* Wo = (const float*)d_in[4];
  const float* bo = (const float*)d_in[5];
  float* out = (float*)d_out;
  char* ws = (char*)d_ws;

  constexpr size_t SZ_X = (size_t)BATCH * SEQ * QD * 2;     // 6,553,600
  constexpr size_t SZ_W = (size_t)QD * INNER * 2;           //   327,680
  constexpr size_t SZ_P = (size_t)BATCH * SEQ * INNER * 2;  // 10,485,760

  bf16_t* xbf = (bf16_t*)(ws);
  bf16_t* WqT = (bf16_t*)(ws + SZ_X);
  bf16_t* WkT = (bf16_t*)(ws + SZ_X + SZ_W);
  bf16_t* WvT = (bf16_t*)(ws + SZ_X + 2 * SZ_W);
  bf16_t* WoT = (bf16_t*)(ws + SZ_X + 3 * SZ_W);
  bf16_t* Qb  = (bf16_t*)(ws + SZ_X + 4 * SZ_W);
  bf16_t* Kb  = (bf16_t*)(ws + SZ_X + 4 * SZ_W + SZ_P);
  bf16_t* Vt  = (bf16_t*)(ws + SZ_X + 4 * SZ_W + 2 * SZ_P);
  bf16_t* Obf = (bf16_t*)(ws + SZ_X + 4 * SZ_W + 3 * SZ_P);
  (void)ws_size; (void)in_sizes; (void)n_in; (void)out_size;

  const int nx = BATCH * SEQ * QD;
  k_cvt<<<(nx + 255) / 256, 256, 0, stream>>>(x, xbf, nx);
  const int nw = QD * INNER;
  k_cvt_tr<<<(nw + 255) / 256, 256, 0, stream>>>(Wq, WqT, QD, INNER);
  k_cvt_tr<<<(nw + 255) / 256, 256, 0, stream>>>(Wk, WkT, QD, INNER);
  k_cvt_tr<<<(nw + 255) / 256, 256, 0, stream>>>(Wv, WvT, QD, INNER);
  k_cvt_tr<<<(nw + 255) / 256, 256, 0, stream>>>(Wo, WoT, INNER, QD);

  dim3 gq(BATCH * SEQ / 128, INNER / 64, 3);
  k_qkv<<<gq, 128, 0, stream>>>(xbf, WqT, WkT, WvT, Qb, Kb, Vt);

  dim3 ga(SEQ / 64, BATCH * HEADS);
  k_attn<<<ga, 128, 0, stream>>>(Qb, Kb, Vt, Obf);

  dim3 go(BATCH * SEQ / 128, QD / 64);
  k_oproj<<<go, 128, 0, stream>>>(Obf, WoT, bo, out);
}
